// MultiHeadAttention_34402688041594
// MI455X (gfx1250) — matrix operand order repaired and checked by reading, not compiled, not run on hardware
//
#include <hip/hip_runtime.h>
#include <math.h>

// ---------------- problem constants ----------------
#define BATCH   2
#define SLEN    2048
#define EMB     1024
#define HID     1024
#define NHEADS  16
#define DK      64
#define SCALE   0.03125f            // 1/sqrt(1024)
#define L2E     1.4426950408889634f

typedef _Float16 h4  __attribute__((ext_vector_type(4)));
typedef _Float16 h8  __attribute__((ext_vector_type(8)));
typedef _Float16 h16 __attribute__((ext_vector_type(16)));
typedef float    f32x4 __attribute__((ext_vector_type(4)));
typedef float    f32x8 __attribute__((ext_vector_type(8)));

static __device__ inline f32x8 wmma16(h16 a, h16 b, f32x8 c) {
  // D = A(16x32 f16) * B(32x16 f16) + C(16x16 f32)
  return __builtin_amdgcn_wmma_f32_16x16x32_f16(false, a, false, b, (short)0, c,
                                                false, false);
}

static __device__ inline h16 cat16(h8 lo, h8 hi) {
  return __builtin_shufflevector(lo, hi, 0,1,2,3,4,5,6,7,8,9,10,11,12,13,14,15);
}

// A-operand fragment: 16 contiguous halfs [p..p+15]
static __device__ inline h16 ldA(const _Float16* p) {
  return cat16(*(const h8*)p, *(const h8*)(p + 8));
}
// B-operand fragment: 16 contiguous halfs [p..p+15]
static __device__ inline h16 ldB(const _Float16* p) {
  return cat16(*(const h8*)p, *(const h8*)(p + 8));
}

// ================================================================
// Kernel 0: one-shot f32 -> f16 conversion (memory-bound pre-pass)
// ================================================================
__global__ __launch_bounds__(256)
void cvt_f32_to_f16(const float* __restrict__ src, _Float16* __restrict__ dst,
                    int n4) {
  const int i = blockIdx.x * 256 + threadIdx.x;
  if (i >= n4) return;
  f32x4 v = ((const f32x4*)src)[i];
  h4 o;
  o[0] = (_Float16)v[0]; o[1] = (_Float16)v[1];
  o[2] = (_Float16)v[2]; o[3] = (_Float16)v[3];
  ((h4*)dst)[i] = o;
}

// ================================================================
// Kernel 1: fused QKV projection.  Y = X @ Wk^T + bk  (same Wk for q,k,v)
//   All-f16 operands, software double-buffered k-loop.
//   q,k outputs -> [B,H,S,DK] f16 ;  v output -> [B,H,DK,S] f16 (transposed)
// ================================================================
__global__ __launch_bounds__(256)
void mha_qkv_proj(const _Float16* __restrict__ q16, const _Float16* __restrict__ k16,
                  const _Float16* __restrict__ v16, const _Float16* __restrict__ Wk16,
                  const float* __restrict__ bk,
                  _Float16* __restrict__ qh, _Float16* __restrict__ kh,
                  _Float16* __restrict__ vT) {
  const int wave = threadIdx.x >> 5;
  const int lane = threadIdx.x & 31;
  const int lmod = lane & 15;
  const int lhi  = lane >> 4;          // 0 or 1
  const int wr = wave >> 2;            // 0..1
  const int wc = wave & 3;             // 0..3
  const int m0 = blockIdx.x * 32 + wr * 16;
  const int n0 = blockIdx.y * 256 + wc * 64;
  const int mode = blockIdx.z;
  const _Float16* X = (mode == 0) ? q16 : (mode == 1) ? k16 : v16;

  const _Float16* ap  = X + (size_t)(m0 + lmod) * EMB + lhi * 16;
  const _Float16* bp0 = Wk16 + (size_t)(n0 +  0 + lmod) * EMB + lhi * 16;
  const _Float16* bp1 = Wk16 + (size_t)(n0 + 16 + lmod) * EMB + lhi * 16;
  const _Float16* bp2 = Wk16 + (size_t)(n0 + 32 + lmod) * EMB + lhi * 16;
  const _Float16* bp3 = Wk16 + (size_t)(n0 + 48 + lmod) * EMB + lhi * 16;

  f32x8 c0 = {}, c1 = {}, c2 = {}, c3 = {};
  h16 a  = ldA(ap);
  h16 b0 = ldB(bp0), b1 = ldB(bp1), b2 = ldB(bp2), b3 = ldB(bp3);

  for (int k0 = 32; k0 < EMB; k0 += 32) {
    // prefetch next fragments (distinct dest VGPRs -> clauseable loads)
    h16 an  = ldA(ap  + k0);
    h16 b0n = ldB(bp0 + k0);
    h16 b1n = ldB(bp1 + k0);
    h16 b2n = ldB(bp2 + k0);
    h16 b3n = ldB(bp3 + k0);
    c0 = wmma16(a, b0, c0);
    c1 = wmma16(a, b1, c1);
    c2 = wmma16(a, b2, c2);
    c3 = wmma16(a, b3, c3);
    a = an; b0 = b0n; b1 = b1n; b2 = b2n; b3 = b3n;
  }
  c0 = wmma16(a, b0, c0);
  c1 = wmma16(a, b1, c1);
  c2 = wmma16(a, b2, c2);
  c3 = wmma16(a, b3, c3);

  f32x8 cc[4] = {c0, c1, c2, c3};
#pragma unroll
  for (int nt = 0; nt < 4; ++nt) {
    const int col = n0 + nt * 16 + lmod;
    const float bias = bk[col];
    const int h = col >> 6;            // head
    const int d = col & 63;            // within-head dim
#pragma unroll
    for (int r = 0; r < 8; ++r) {
      const int row = m0 + r + lhi * 8;          // global token row (0..4095)
      const int b = row >> 11;                   // row / SLEN
      const int s = row & (SLEN - 1);
      const float v = cc[nt][r] + bias;
      if (mode == 0)      qh[(((b * NHEADS + h) * SLEN) + s) * DK + d] = (_Float16)v;
      else if (mode == 1) kh[(((b * NHEADS + h) * SLEN) + s) * DK + d] = (_Float16)v;
      else                vT[(((b * NHEADS + h) * DK) + d) * SLEN + s] = (_Float16)v;
    }
  }
}

// ================================================================
// Kernel 2: flash attention, one wave per (b, h, 16-query block).
//   V-tile loads issued early so they overlap softmax + LDS bounce.
// ================================================================
__global__ __launch_bounds__(32)
void mha_flash_attn(const _Float16* __restrict__ qh, const _Float16* __restrict__ kh,
                    const _Float16* __restrict__ vT, _Float16* __restrict__ concat) {
  __shared__ __align__(16) _Float16 sP[16 * 32];   // staged P tile (A-layout bounce)

  const int lane = threadIdx.x & 31;
  const int lmod = lane & 15;
  const int lhi  = lane >> 4;
  const int s0 = blockIdx.x * 16;
  const int h  = blockIdx.y;
  const int b  = blockIdx.z;

  const _Float16* qbase = qh + (size_t)(b * NHEADS + h) * SLEN * DK;
  const _Float16* kbase = kh + (size_t)(b * NHEADS + h) * SLEN * DK;
  const _Float16* vbase = vT + (size_t)(b * NHEADS + h) * DK * SLEN;

  // Q A-fragments for d-chunks 0..31 and 32..63 (held for the whole pass)
  h16 qa0 = ldA(qbase + (s0 + lmod) * DK +  0 + lhi * 16);
  h16 qa1 = ldA(qbase + (s0 + lmod) * DK + 32 + lhi * 16);

  f32x8 ctx0 = {}, ctx1 = {}, ctx2 = {}, ctx3 = {};
  float m_i[8], l_i[8];
#pragma unroll
  for (int r = 0; r < 8; ++r) { m_i[r] = -__builtin_inff(); l_i[r] = 0.0f; }

  for (int t0 = 0; t0 < SLEN; t0 += 32) {
    // ---- K tiles (score B-operands) + V tiles (ctx B-operands) up front ----
    const _Float16* kr0 = kbase + (t0 +  0 + lmod) * DK + lhi * 16;
    const _Float16* kr1 = kbase + (t0 + 16 + lmod) * DK + lhi * 16;
    h16 kb00 = ldB(kr0);        // d 0..31  for keys t0..t0+15
    h16 kb01 = ldB(kr0 + 32);   // d 32..63
    h16 kb10 = ldB(kr1);        // d 0..31  for keys t0+16..t0+31
    h16 kb11 = ldB(kr1 + 32);   // d 32..63
    h16 vb0 = ldB(vbase + ( 0 + lmod) * SLEN + t0 + lhi * 16);
    h16 vb1 = ldB(vbase + (16 + lmod) * SLEN + t0 + lhi * 16);
    h16 vb2 = ldB(vbase + (32 + lmod) * SLEN + t0 + lhi * 16);
    h16 vb3 = ldB(vbase + (48 + lmod) * SLEN + t0 + lhi * 16);

    // ---- scores: two 16x16 tiles covering keys t0..t0+31 ----
    f32x8 sc0 = {}, sc1 = {};
    sc0 = wmma16(qa0, kb00, sc0);
    sc0 = wmma16(qa1, kb01, sc0);
    sc1 = wmma16(qa0, kb10, sc1);
    sc1 = wmma16(qa1, kb11, sc1);

    // ---- online softmax (each lane owns stats for its 8 C-layout rows) ----
    float alpha[8];
#pragma unroll
    for (int r = 0; r < 8; ++r) {
      const float a0 = sc0[r] * SCALE;
      const float a1 = sc1[r] * SCALE;
      float mx = fmaxf(a0, a1);
      mx = fmaxf(mx, __shfl_xor(mx, 1, 16));
      mx = fmaxf(mx, __shfl_xor(mx, 2, 16));
      mx = fmaxf(mx, __shfl_xor(mx, 4, 16));
      mx = fmaxf(mx, __shfl_xor(mx, 8, 16));
      const float mnew = fmaxf(m_i[r], mx);
      alpha[r] = exp2f((m_i[r] - mnew) * L2E);
      m_i[r] = mnew;
      const float p0 = exp2f((a0 - mnew) * L2E);
      const float p1 = exp2f((a1 - mnew) * L2E);
      float rs = p0 + p1;
      rs += __shfl_xor(rs, 1, 16);
      rs += __shfl_xor(rs, 2, 16);
      rs += __shfl_xor(rs, 4, 16);
      rs += __shfl_xor(rs, 8, 16);
      l_i[r] = l_i[r] * alpha[r] + rs;
      // stage P (f16) for the A-operand bounce
      const int row = lhi * 8 + r;
      sP[row * 32 +      lmod] = (_Float16)p0;
      sP[row * 32 + 16 + lmod] = (_Float16)p1;
    }

    // rescale running context by alpha (same row<->VGPR mapping as C layout)
#pragma unroll
    for (int r = 0; r < 8; ++r) {
      ctx0[r] *= alpha[r]; ctx1[r] *= alpha[r];
      ctx2[r] *= alpha[r]; ctx3[r] *= alpha[r];
    }

    // same-wave DS is in-order; barrier is for the compiler scheduler
    asm volatile("s_wait_dscnt 0" ::: "memory");

    // ---- P (16x32) read back in A-operand layout, then ctx += P @ V ----
    h16 pa = ldA(&sP[lmod * 32 + lhi * 16]);
    ctx0 = wmma16(pa, vb0, ctx0);
    ctx1 = wmma16(pa, vb1, ctx1);
    ctx2 = wmma16(pa, vb2, ctx2);
    ctx3 = wmma16(pa, vb3, ctx3);
    asm volatile("" ::: "memory");
  }

  // ---- normalize and write concat[b, s, h*64 + d] (f16) ----
  f32x8 cx[4] = {ctx0, ctx1, ctx2, ctx3};
#pragma unroll
  for (int r = 0; r < 8; ++r) {
    const float inv = 1.0f / l_i[r];
    const int row = s0 + r + lhi * 8;
#pragma unroll
    for (int dt = 0; dt < 4; ++dt) {
      const int d = dt * 16 + lmod;
      concat[((size_t)(b * SLEN + row)) * HID + h * DK + d] = (_Float16)(cx[dt][r] * inv);
    }
  }
}

// ================================================================
// Kernel 3: output projection.  out = concat @ Wo^T + bo   (f32 out)
// ================================================================
__global__ __launch_bounds__(256)
void mha_out_proj(const _Float16* __restrict__ concat, const _Float16* __restrict__ Wo16,
                  const float* __restrict__ bo, float* __restrict__ out) {
  const int wave = threadIdx.x >> 5;
  const int lane = threadIdx.x & 31;
  const int lmod = lane & 15;
  const int lhi  = lane >> 4;
  const int wr = wave >> 2;
  const int wc = wave & 3;
  const int m0 = blockIdx.x * 32 + wr * 16;
  const int n0 = blockIdx.y * 256 + wc * 64;

  const _Float16* ap  = concat + (size_t)(m0 + lmod) * HID + lhi * 16;
  const _Float16* bp0 = Wo16 + (size_t)(n0 +  0 + lmod) * HID + lhi * 16;
  const _Float16* bp1 = Wo16 + (size_t)(n0 + 16 + lmod) * HID + lhi * 16;
  const _Float16* bp2 = Wo16 + (size_t)(n0 + 32 + lmod) * HID + lhi * 16;
  const _Float16* bp3 = Wo16 + (size_t)(n0 + 48 + lmod) * HID + lhi * 16;

  f32x8 c0 = {}, c1 = {}, c2 = {}, c3 = {};
  h16 a  = ldA(ap);
  h16 b0 = ldB(bp0), b1 = ldB(bp1), b2 = ldB(bp2), b3 = ldB(bp3);

  for (int k0 = 32; k0 < HID; k0 += 32) {
    h16 an  = ldA(ap  + k0);
    h16 b0n = ldB(bp0 + k0);
    h16 b1n = ldB(bp1 + k0);
    h16 b2n = ldB(bp2 + k0);
    h16 b3n = ldB(bp3 + k0);
    c0 = wmma16(a, b0, c0);
    c1 = wmma16(a, b1, c1);
    c2 = wmma16(a, b2, c2);
    c3 = wmma16(a, b3, c3);
    a = an; b0 = b0n; b1 = b1n; b2 = b2n; b3 = b3n;
  }
  c0 = wmma16(a, b0, c0);
  c1 = wmma16(a, b1, c1);
  c2 = wmma16(a, b2, c2);
  c3 = wmma16(a, b3, c3);

  f32x8 cc[4] = {c0, c1, c2, c3};
#pragma unroll
  for (int nt = 0; nt < 4; ++nt) {
    const int col = n0 + nt * 16 + lmod;
    const float bias = bo[col];
#pragma unroll
    for (int r = 0; r < 8; ++r) {
      const int row = m0 + r + lhi * 8;
      out[(size_t)row * EMB + col] = cc[nt][r] + bias;
    }
  }
}

// ================================================================
// launch
// ================================================================
extern "C" void kernel_launch(void* const* d_in, const int* in_sizes, int n_in,
                              void* d_out, int out_size, void* d_ws, size_t ws_size,
                              hipStream_t stream) {
  const float* q  = (const float*)d_in[0];
  const float* k  = (const float*)d_in[1];
  const float* v  = (const float*)d_in[2];
  const float* Wk = (const float*)d_in[3];
  const float* bk = (const float*)d_in[4];
  const float* Wo = (const float*)d_in[5];
  const float* bo = (const float*)d_in[6];
  float* out = (float*)d_out;

  // workspace layout (f16), ~60 MB total:
  //   qh | kh | vT | concat : 4 * 4,194,304 halfs
  //   q16 | k16 | v16       : 3 * 4,194,304 halfs
  //   Wk16 | Wo16           : 2 * 1,048,576 halfs
  const size_t HEADELEMS = (size_t)BATCH * NHEADS * SLEN * DK;   // 4,194,304
  const size_t WELEMS    = (size_t)HID * EMB;                    // 1,048,576
  _Float16* qh     = (_Float16*)d_ws;
  _Float16* kh     = qh + HEADELEMS;
  _Float16* vT     = kh + HEADELEMS;
  _Float16* concat = vT + HEADELEMS;
  _Float16* q16    = concat + HEADELEMS;
  _Float16* k16    = q16 + HEADELEMS;
  _Float16* v16    = k16 + HEADELEMS;
  _Float16* Wk16   = v16 + HEADELEMS;
  _Float16* Wo16   = Wk16 + WELEMS;

  // 0) f32 -> f16 pre-pass (activations + weights)
  const int n4_act = (int)(HEADELEMS / 4);   // 1,048,576 float4 groups
  const int n4_w   = (int)(WELEMS / 4);      //   262,144
  cvt_f32_to_f16<<<dim3((n4_act + 255) / 256), dim3(256), 0, stream>>>(q, q16, n4_act);
  cvt_f32_to_f16<<<dim3((n4_act + 255) / 256), dim3(256), 0, stream>>>(k, k16, n4_act);
  cvt_f32_to_f16<<<dim3((n4_act + 255) / 256), dim3(256), 0, stream>>>(v, v16, n4_act);
  cvt_f32_to_f16<<<dim3((n4_w + 255) / 256), dim3(256), 0, stream>>>(Wk, Wk16, n4_w);
  cvt_f32_to_f16<<<dim3((n4_w + 255) / 256), dim3(256), 0, stream>>>(Wo, Wo16, n4_w);

  // 1) QKV projection: M=4096 rows, N=1024 cols, z in {q,k,v}
  dim3 g1((BATCH * SLEN) / 32, HID / 256, 3);
  mha_qkv_proj<<<g1, dim3(256, 1, 1), 0, stream>>>(q16, k16, v16, Wk16, bk, qh, kh, vT);

  // 2) flash attention: (query-block, head, batch), one wave each
  dim3 g2(SLEN / 16, NHEADS, BATCH);
  mha_flash_attn<<<g2, dim3(32, 1, 1), 0, stream>>>(qh, kh, vT, concat);

  // 3) output projection
  dim3 g3((BATCH * SLEN) / 32, EMB / 256, 1);
  mha_out_proj<<<g3, dim3(256, 1, 1), 0, stream>>>(concat, Wo16, bo, out);
}
